// CrossCBR_3710851743761
// MI455X (gfx1250) — matrix operand order repaired and checked by reading, not compiled, not run on hardware
//
#include <hip/hip_runtime.h>
#include <hip/hip_fp16.h>

typedef __attribute__((ext_vector_type(16))) _Float16 v16h;
typedef __attribute__((ext_vector_type(8)))  float    v8f;

#define THREADS 256

// ---------------- utility kernels ----------------

__global__ void k_zero(float* __restrict__ p, long n) {
  long i = (long)blockIdx.x * blockDim.x + threadIdx.x;
  if (i < n) p[i] = 0.0f;
}

// f = concat(A, B); tot = f  (element counts are rows*64)
__global__ void k_concat(const float* __restrict__ A, const float* __restrict__ Bf,
                         float* __restrict__ f, float* __restrict__ tot,
                         long nAe, long nBe) {
  long i = (long)blockIdx.x * blockDim.x + threadIdx.x;
  if (i < nAe)            { float x = A[i];        f[i] = x; tot[i] = x; }
  else if (i < nAe + nBe) { float x = Bf[i - nAe]; f[i] = x; tot[i] = x; }
}

// fout[r[e]] += scale * v[e] * fin[c[e]]  (D=64, 16 threads/edge, float4 gather)
__global__ void k_spmm(const int* __restrict__ r, const int* __restrict__ c,
                       const float* __restrict__ v, const float* __restrict__ fin,
                       float* __restrict__ fout, long nnz, float scale) {
  long t = (long)blockIdx.x * blockDim.x + threadIdx.x;
  long e = t >> 4;
  if (e >= nnz) return;
  int q = (int)(t & 15) * 4;
  const float4 s = *(const float4*)(fin + (long)c[e] * 64 + q);
  float* dst = fout + (long)r[e] * 64 + q;
  float vv = v[e] * scale;
  atomicAdd(dst + 0, vv * s.x);
  atomicAdd(dst + 1, vv * s.y);
  atomicAdd(dst + 2, vv * s.z);
  atomicAdd(dst + 3, vv * s.w);
}

// tot[row] += l2norm(f[row]); one wave32 per row, 2 dims/lane
__global__ void k_l2acc(const float* __restrict__ f, float* __restrict__ tot, long n) {
  long wave = ((long)blockIdx.x * blockDim.x + threadIdx.x) >> 5;
  int lane = threadIdx.x & 31;
  if (wave >= n) return;
  const float* row = f + wave * 64;
  int d = lane * 2;
  float x0 = row[d], x1 = row[d + 1];
  float s = x0 * x0 + x1 * x1;
  #pragma unroll
  for (int m = 16; m; m >>= 1) s += __shfl_xor(s, m, 32);
  float inv = 1.0f / fmaxf(sqrtf(s), 1e-12f);
  tot[wave * 64 + d]     += x0 * inv;
  tot[wave * 64 + d + 1] += x1 * inv;
}

// BPR: one wave per batch element
__global__ void k_bpr(const int* __restrict__ users, const int* __restrict__ bundles,
                      const float* __restrict__ il_u, const float* __restrict__ il_b,
                      const float* __restrict__ bl, long U, float* __restrict__ acc, int BS) {
  int w = (int)(((long)blockIdx.x * blockDim.x + threadIdx.x) >> 5);
  int lane = threadIdx.x & 31;
  if (w >= BS) return;
  int u  = users[w];
  int b0 = bundles[2 * w], b1 = bundles[2 * w + 1];
  const float* iu  = il_u + (long)u * 64;
  const float* bu  = bl   + (long)u * 64;
  const float* ib0 = il_b + (long)b0 * 64;
  const float* ib1 = il_b + (long)b1 * 64;
  const float* bb0 = bl + (U + (long)b0) * 64;
  const float* bb1 = bl + (U + (long)b1) * 64;
  int d = lane * 2;
  float p0 = iu[d]*ib0[d] + iu[d+1]*ib0[d+1] + bu[d]*bb0[d] + bu[d+1]*bb0[d+1];
  float p1 = iu[d]*ib1[d] + iu[d+1]*ib1[d+1] + bu[d]*bb1[d] + bu[d+1]*bb1[d+1];
  #pragma unroll
  for (int m = 16; m; m >>= 1) { p0 += __shfl_xor(p0, m, 32); p1 += __shfl_xor(p1, m, 32); }
  if (lane == 0) {
    float x = p0 - p1;                 // -log_sigmoid(x) = softplus(-x)
    float l = (x > 0.f) ? log1pf(expf(-x)) : (-x + log1pf(expf(x)));
    atomicAdd(acc + 0, l / (float)BS);
  }
}

// gather rows, l2-normalize in f32, pack to f16, ps[i] = dot(posN,augN)/C_TEMP
__global__ void k_gathnorm(const float* __restrict__ srcA, const float* __restrict__ srcB,
                           const int* __restrict__ idx, int stride, long offA, long offB,
                           __half* __restrict__ outA, __half* __restrict__ outB,
                           float* __restrict__ ps, int n, float invT) {
  int w = (int)(((long)blockIdx.x * blockDim.x + threadIdx.x) >> 5);
  int lane = threadIdx.x & 31;
  if (w >= n) return;
  int j = idx[(long)w * stride];
  const float* ra = srcA + (offA + (long)j) * 64;
  const float* rb = srcB + (offB + (long)j) * 64;
  int d = lane * 2;
  float a0 = ra[d], a1 = ra[d + 1], b0 = rb[d], b1 = rb[d + 1];
  float na = a0 * a0 + a1 * a1, nb = b0 * b0 + b1 * b1;
  #pragma unroll
  for (int m = 16; m; m >>= 1) { na += __shfl_xor(na, m, 32); nb += __shfl_xor(nb, m, 32); }
  float ia = 1.0f / fmaxf(sqrtf(na), 1e-12f);
  float ib = 1.0f / fmaxf(sqrtf(nb), 1e-12f);
  float an0 = a0 * ia, an1 = a1 * ia, bn0 = b0 * ib, bn1 = b1 * ib;
  outA[(long)w * 64 + d]     = __float2half(an0);
  outA[(long)w * 64 + d + 1] = __float2half(an1);
  outB[(long)w * 64 + d]     = __float2half(bn0);
  outB[(long)w * 64 + d + 1] = __float2half(bn1);
  float p = an0 * bn0 + an1 * bn1;
  #pragma unroll
  for (int m = 16; m; m >>= 1) p += __shfl_xor(p, m, 32);
  if (lane == 0) ps[w] = p * invT;
}

// tt = (pos @ aug^T) * invT via v_wmma_f32_16x16x32_f16; one wave per 16x16 tile
__global__ void k_ttgemm(const __half* __restrict__ Ph, const __half* __restrict__ Qh,
                         float* __restrict__ tt, int N, float invT) {
  const _Float16* P = reinterpret_cast<const _Float16*>(Ph);
  const _Float16* Q = reinterpret_cast<const _Float16*>(Qh);
  int wave = (int)(((long)blockIdx.x * blockDim.x + threadIdx.x) >> 5);
  int lane = threadIdx.x & 31;
  int tilesN = N >> 4;
  int tm = wave / tilesN, tn = wave % tilesN;
  if (tm >= tilesN) return;
  int mrow = tm * 16 + (lane & 15);
  int ncol = tn * 16 + (lane & 15);
  int hiA = (lane >> 4) * 16;   // A: lanes 0-15 hold K 0-15, lanes 16-31 K 16-31
  int hiB = (lane >> 4) * 16;   // B: lanes 0-15 hold K 0-15, lanes 16-31 K 16-31
  v8f acc = {};
  #pragma unroll
  for (int ks = 0; ks < 64; ks += 32) {
    v16h a, b;
    const _Float16* ap = P + (long)mrow * 64 + ks + hiA;
    #pragma unroll
    for (int j = 0; j < 8; ++j) { a[j] = ap[j]; a[8 + j] = ap[8 + j]; }
    const _Float16* bp = Q + (long)ncol * 64 + ks + hiB;
    #pragma unroll
    for (int j = 0; j < 16; ++j) b[j] = bp[j];
    acc = __builtin_amdgcn_wmma_f32_16x16x32_f16(false, a, false, b,
                                                 (short)0, acc, false, false);
  }
  int rbase = tm * 16 + (lane >> 4) * 8;   // C/D: VGPR r -> row r (lanes 0-15) / 8+r (16-31)
  int col = tn * 16 + (lane & 15);
  #pragma unroll
  for (int r2 = 0; r2 < 8; ++r2)
    tt[(long)(rbase + r2) * N + col] = acc[r2] * invT;
}

// row-wise logsumexp; acc[1] += (lse - ps[row]) * wscale
__global__ void k_lse(const float* __restrict__ tt, const float* __restrict__ ps,
                      float* __restrict__ acc, int N, float wscale) {
  __shared__ float sh[THREADS];
  int row = blockIdx.x;
  const float* r = tt + (long)row * N;
  float mx = -3.4e38f;
  for (int c = threadIdx.x; c < N; c += blockDim.x) mx = fmaxf(mx, r[c]);
  sh[threadIdx.x] = mx; __syncthreads();
  for (int s = blockDim.x / 2; s; s >>= 1) {
    if ((int)threadIdx.x < s) sh[threadIdx.x] = fmaxf(sh[threadIdx.x], sh[threadIdx.x + s]);
    __syncthreads();
  }
  mx = sh[0]; __syncthreads();
  float sum = 0.0f;
  for (int c = threadIdx.x; c < N; c += blockDim.x) sum += expf(r[c] - mx);
  sh[threadIdx.x] = sum; __syncthreads();
  for (int s = blockDim.x / 2; s; s >>= 1) {
    if ((int)threadIdx.x < s) sh[threadIdx.x] += sh[threadIdx.x + s];
    __syncthreads();
  }
  if (threadIdx.x == 0) {
    float lse = mx + logf(sh[0]);
    atomicAdd(acc + 1, (lse - ps[row]) * wscale);
  }
}

__global__ void k_final(const float* __restrict__ acc, float* __restrict__ out) {
  if (threadIdx.x == 0) { out[0] = acc[0]; out[1] = acc[1]; }
}

// ---------------- host ----------------

static inline long cdiv(long a, long b) { return (a + b - 1) / b; }

extern "C" void kernel_launch(void* const* d_in, const int* in_sizes, int n_in,
                              void* d_out, int out_size, void* d_ws, size_t ws_size,
                              hipStream_t stream) {
  const int*   users   = (const int*)d_in[0];
  const int*   bundles = (const int*)d_in[1];
  const float* uf      = (const float*)d_in[2];
  const float* bf      = (const float*)d_in[3];
  const float* itf     = (const float*)d_in[4];
  const int*   ilg_r   = (const int*)d_in[5];
  const int*   ilg_c   = (const int*)d_in[6];
  const float* ilg_v   = (const float*)d_in[7];
  const int*   blg_r   = (const int*)d_in[8];
  const int*   blg_c   = (const int*)d_in[9];
  const float* blg_v   = (const float*)d_in[10];
  const int*   bag_r   = (const int*)d_in[11];
  const int*   bag_c   = (const int*)d_in[12];
  const float* bag_v   = (const float*)d_in[13];

  const long D  = 64;
  const long U  = in_sizes[2] / D;
  const long Bn = in_sizes[3] / D;
  const long In = in_sizes[4] / D;
  const int  BS = in_sizes[0];
  const long nnz_il  = in_sizes[5];
  const long nnz_bl  = in_sizes[8];
  const long nnz_bag = in_sizes[11];
  const long n1 = U + In, n2 = U + Bn;
  const float invT = 4.0f;  // 1/C_TEMP

  // workspace carve-up (floats)
  float* fA     = (float*)d_ws;         // n1*64  (tt aliases here later)
  float* fB     = fA + n1 * D;          // n1*64
  float* tot_il = fB + n1 * D;          // n1*64
  float* tot_bl = tot_il + n1 * D;      // n2*64
  float* il_b   = tot_bl + n2 * D;      // Bn*64
  float* ps     = il_b + Bn * D;        // BS
  float* acc    = ps + BS;              // 8
  __half* posH  = (__half*)(acc + 8);   // BS*64
  __half* augH  = posH + (long)BS * D;  // BS*64
  float* tt     = fA;                   // BS*BS  (<= n1*64, reuse after propagation)

  k_zero<<<1, 8, 0, stream>>>(acc, 8);

  // ---- item-level propagation over U+I graph ----
  {
    long ne = n1 * D;
    k_concat<<<(unsigned)cdiv(ne, THREADS), THREADS, 0, stream>>>(uf, itf, fA, tot_il, U * D, In * D);
    float* cur = fA; float* nxt = fB;
    for (int layer = 0; layer < 2; ++layer) {
      k_zero<<<(unsigned)cdiv(ne, THREADS), THREADS, 0, stream>>>(nxt, ne);
      long t = nnz_il * 16;
      k_spmm<<<(unsigned)cdiv(t, THREADS), THREADS, 0, stream>>>(ilg_r, ilg_c, ilg_v, cur, nxt, nnz_il, 1.0f / (float)(layer + 2));
      k_l2acc<<<(unsigned)cdiv(n1 * 32, THREADS), THREADS, 0, stream>>>(nxt, tot_il, n1);
      float* tmp = cur; cur = nxt; nxt = tmp;
    }
  }

  // IL_b = bag-aggregation of IL_i
  k_zero<<<(unsigned)cdiv(Bn * D, THREADS), THREADS, 0, stream>>>(il_b, Bn * D);
  {
    long t = nnz_bag * 16;
    k_spmm<<<(unsigned)cdiv(t, THREADS), THREADS, 0, stream>>>(bag_r, bag_c, bag_v, tot_il + U * D, il_b, nnz_bag, 1.0f);
  }

  // ---- bundle-level propagation over U+B graph ----
  {
    long ne = n2 * D;
    k_concat<<<(unsigned)cdiv(ne, THREADS), THREADS, 0, stream>>>(uf, bf, fA, tot_bl, U * D, Bn * D);
    float* cur = fA; float* nxt = fB;
    for (int layer = 0; layer < 2; ++layer) {
      k_zero<<<(unsigned)cdiv(ne, THREADS), THREADS, 0, stream>>>(nxt, ne);
      long t = nnz_bl * 16;
      k_spmm<<<(unsigned)cdiv(t, THREADS), THREADS, 0, stream>>>(blg_r, blg_c, blg_v, cur, nxt, nnz_bl, 1.0f / (float)(layer + 2));
      k_l2acc<<<(unsigned)cdiv(n2 * 32, THREADS), THREADS, 0, stream>>>(nxt, tot_bl, n2);
      float* tmp = cur; cur = nxt; nxt = tmp;
    }
  }

  // ---- BPR loss ----
  k_bpr<<<(unsigned)cdiv((long)BS * 32, THREADS), THREADS, 0, stream>>>(users, bundles, tot_il, il_b, tot_bl, U, acc, BS);

  // ---- contrastive loss: users ----
  const float wscale = 1.0f / (2.0f * (float)BS);
  {
    k_gathnorm<<<(unsigned)cdiv((long)BS * 32, THREADS), THREADS, 0, stream>>>(
        tot_il, tot_bl, users, 1, 0, 0, posH, augH, ps, BS, invT);
    long waves = (long)(BS / 16) * (BS / 16);
    k_ttgemm<<<(unsigned)cdiv(waves * 32, THREADS), THREADS, 0, stream>>>(posH, augH, tt, BS, invT);
    k_lse<<<BS, THREADS, 0, stream>>>(tt, ps, acc, BS, wscale);
  }

  // ---- contrastive loss: bundles ----
  {
    k_gathnorm<<<(unsigned)cdiv((long)BS * 32, THREADS), THREADS, 0, stream>>>(
        il_b, tot_bl, bundles, 2, 0, U, posH, augH, ps, BS, invT);
    long waves = (long)(BS / 16) * (BS / 16);
    k_ttgemm<<<(unsigned)cdiv(waves * 32, THREADS), THREADS, 0, stream>>>(posH, augH, tt, BS, invT);
    k_lse<<<BS, THREADS, 0, stream>>>(tt, ps, acc, BS, wscale);
  }

  k_final<<<1, 32, 0, stream>>>(acc, (float*)d_out);
}
